// MetaGraph2_31044023616141
// MI455X (gfx1250) — matrix operand order repaired and checked by reading, not compiled, not run on hardware
//
#include <hip/hip_runtime.h>
#include <hip/hip_bf16.h>
#include <math.h>

typedef __attribute__((ext_vector_type(16))) _Float16 v16h;
typedef __attribute__((ext_vector_type(8)))  _Float16 v8h;
typedef __attribute__((ext_vector_type(8)))  float    v8f;

#define NHEAD  8
#define EMB    12
#define DPAD   16
#define HIDDEN 32
#define TOPK   5
#define B_     2
#define N_     2048
#define T_     288
#define DX     145          // T/2+1
#define FPAD   160          // DX padded to 10 WMMA col-tiles
#define DY     408          // 12*HIDDEN + 2*EMB

// ---------------------------------------------------------------- utilities
// min-max + l2 normalize a length-n segment (blockDim must be 256)
__device__ void norm_seg(const float* src, int n, float* dst, int tid, float* red) {
  float mn = 3.4e38f, mx = -3.4e38f;
  for (int i = tid; i < n; i += 256) { float v = src[i]; mn = fminf(mn, v); mx = fmaxf(mx, v); }
  red[tid] = mn; __syncthreads();
  for (int s = 128; s > 0; s >>= 1) { if (tid < s) red[tid] = fminf(red[tid], red[tid + s]); __syncthreads(); }
  mn = red[0]; __syncthreads();
  red[tid] = mx; __syncthreads();
  for (int s = 128; s > 0; s >>= 1) { if (tid < s) red[tid] = fmaxf(red[tid], red[tid + s]); __syncthreads(); }
  mx = red[0]; __syncthreads();
  float ss = 0.f;
  float inv = 1.0f / (mx - mn + 1.0f);
  for (int i = tid; i < n; i += 256) { float v = (src[i] - mn) * inv; ss += v * v; }
  red[tid] = ss; __syncthreads();
  for (int s = 128; s > 0; s >>= 1) { if (tid < s) red[tid] += red[tid + s]; __syncthreads(); }
  float nrm = fmaxf(sqrtf(red[0]), 1e-12f); __syncthreads();
  float rn = 1.0f / nrm;
  for (int i = tid; i < n; i += 256) dst[i] = (src[i] - mn) * inv * rn;
}

// -------------------------------------------- K0a: x -> f16 copy for WMMA DFT
__global__ __launch_bounds__(256)
void k_tof16(const float* __restrict__ x, _Float16* __restrict__ xh, int n) {
  int i = blockIdx.x * 256 + threadIdx.x;
  if (i < n) xh[i] = (_Float16)x[i];
}

// ---------------------- K0b: twiddle tables (transposed, f-major) in f16
__global__ __launch_bounds__(256)
void k_twiddle(_Float16* __restrict__ WcT, _Float16* __restrict__ WsT) {
  int i = blockIdx.x * 256 + threadIdx.x;      // f*T + t
  if (i >= FPAD * T_) return;
  int f = i / T_, t = i % T_;
  int ph = (f * t) % T_;
  float ang = -6.283185307179586f * (float)ph / (float)T_;
  float s, c; __sincosf(ang, &s, &c);
  WcT[i] = (_Float16)c; WsT[i] = (_Float16)s;
}

// ------- K1: DFT magnitude as WMMA GEMM: mag = |X @ Wdft| (K = 288 = 9x32)
__global__ __launch_bounds__(32)
void k_dft(const _Float16* __restrict__ xh, const _Float16* __restrict__ WcT,
           const _Float16* __restrict__ WsT, float* __restrict__ mag) {
  int wg = blockIdx.x;                 // rtile*10 + ftile
  int ftile = wg % (FPAD / 16), rtile = wg / (FPAD / 16);
  int lane = threadIdx.x;
  int half = lane >> 4, lm = lane & 15;
  const _Float16* Xr = xh  + (size_t)(rtile * 16 + lm) * T_;      // A row lm
  const _Float16* Ct = WcT + (size_t)(ftile * 16 + lm) * T_;      // B col lm
  const _Float16* St = WsT + (size_t)(ftile * 16 + lm) * T_;
  v8f oc = {}, os = {};
  for (int kb = 0; kb < T_; kb += 32) {
    if (kb + 32 < T_) __builtin_prefetch(Xr + kb + 32, 0, 3);
    v16h a;
    {
      v8h lo = *(const v8h*)(Xr + kb + 16 * half);
      v8h hi = *(const v8h*)(Xr + kb + 16 * half + 8);
      #pragma unroll
      for (int e = 0; e < 8; ++e) { a[e] = lo[e]; a[8 + e] = hi[e]; }
    }
    v16h bc = *(const v16h*)(Ct + kb + 16 * half);
    v16h bs = *(const v16h*)(St + kb + 16 * half);
    oc = __builtin_amdgcn_wmma_f32_16x16x32_f16(false, a, false, bc, (short)0, oc, false, false);
    os = __builtin_amdgcn_wmma_f32_16x16x32_f16(false, a, false, bs, (short)0, os, false, false);
  }
  #pragma unroll
  for (int r = 0; r < 8; ++r) {
    int m = 8 * half + r;
    mag[(size_t)(rtile * 16 + m) * FPAD + ftile * 16 + lm] =
        sqrtf(oc[r] * oc[r] + os[r] * os[r]) * 0.05892556509887896f; // 1/sqrt(288)
  }
}

// --------------------------------- K1b: min-max + l2 normalize spectral rows
__global__ __launch_bounds__(256)
void k_specnorm(const float* __restrict__ mag, float* __restrict__ xn) {
  __shared__ float red[256];
  int row = blockIdx.x;
  norm_seg(mag + (size_t)row * FPAD, DX, xn + (size_t)row * DX, threadIdx.x, red);
}

// ------------------------------------------------- K2: [tdn, locn] concat
__global__ __launch_bounds__(256)
void k_tdloc(const float* __restrict__ loc, const float* __restrict__ td,
             float* __restrict__ tl) {
  int row = blockIdx.x;
  int tid = threadIdx.x;
  __shared__ float red[256];
  norm_seg(td  + (size_t)row * (2 * EMB),     2 * EMB,     tl + (size_t)row * DY,            tid, red);
  __syncthreads();
  norm_seg(loc + (size_t)row * (12 * HIDDEN), 12 * HIDDEN, tl + (size_t)row * DY + 2 * EMB,  tid, red);
}

// ---------------------------- K3: fold proj + head weights: C = wa @ Wb
__global__ __launch_bounds__(256)
void k_combine(const float* __restrict__ wa, const float* __restrict__ wb,
               float* __restrict__ out, int rows) {
  int i = blockIdx.x * blockDim.x + threadIdx.x;
  if (i >= rows * 96) return;
  int r = i / 96, c = i % 96;
  float acc = 0.f;
  for (int e = 0; e < EMB; ++e) acc += wa[r * EMB + e] * wb[e * 96 + c];
  out[i] = acc;
}

// ---- K4: q/k heads f16 padded [B*H][N][16]; V transposed [B*H][16][N]
//      (q pre-scaled by 1/sqrt(EMB) so the attention loop is scale-free)
__global__ __launch_bounds__(128)
void k_qkv(const float* __restrict__ tl, const float* __restrict__ xn,
           const float* __restrict__ Cq, const float* __restrict__ Ck,
           const float* __restrict__ Cv,
           _Float16* __restrict__ qh, _Float16* __restrict__ kh, _Float16* __restrict__ vhT) {
  int row = blockIdx.x;               // b*N + n
  int b = row / N_, n = row % N_;
  int tid = threadIdx.x;
  __shared__ float a[DY];
  __shared__ float xrow[DX];
  for (int i = tid; i < DY; i += 128) a[i]    = tl[(size_t)row * DY + i];
  for (int i = tid; i < DX; i += 128) xrow[i] = xn[(size_t)row * DX + i];
  __syncthreads();
  int h = tid >> 4, d = tid & 15;     // padded output slot
  float q = 0.f, k = 0.f, v = 0.f;
  if (d < EMB) {
    int c = h * EMB + d;
    for (int i = 0; i < DY; ++i) q += a[i] * Cq[i * 96 + c];
    for (int i = 0; i < DX; ++i) { float xi = xrow[i]; k += xi * Ck[i * 96 + c]; v += xi * Cv[i * 96 + c]; }
    q *= 0.28867513459481287f;        // 1/sqrt(EMB)
  }
  size_t bh = (size_t)b * NHEAD + h;
  qh [(bh * N_ + n) * DPAD + d] = (_Float16)q;
  kh [(bh * N_ + n) * DPAD + d] = (_Float16)k;
  vhT[(bh * DPAD + d) * N_ + n] = (_Float16)v;
}

// ---------------- K5: flash attention, 1 wave per (b,h,16-row q tile)
__global__ __launch_bounds__(32)
void k_attn(const _Float16* __restrict__ qh, const _Float16* __restrict__ kh,
            const _Float16* __restrict__ vhT, float* __restrict__ attn_out) {
  int wg = blockIdx.x;                // (b*NHEAD+h)*128 + qt
  int qt = wg & 127;
  int bh = wg >> 7;
  int b = bh >> 3, h = bh & 7;
  int lane = threadIdx.x;
  int half = lane >> 4, lm = lane & 15;
  const _Float16* Qb  = qh  + (((size_t)bh) * N_ + (size_t)qt * 16) * DPAD;
  const _Float16* Kb  = kh  + ((size_t)bh) * N_ * DPAD;
  const _Float16* VbT = vhT + ((size_t)bh) * DPAD * N_ + (size_t)lm * N_;  // emb row lm

  // Q A-fragment (contiguous k): half 0 = Q[lm][0..15], half 1 zero pad
  v16h aq = {};
  {
    if (half == 0) aq = *(const v16h*)(Qb + lm * DPAD);
  }

  v8f o = {};
  float rmax = -3.4e38f, rsum = 0.f;  // valid on lanes 0..15 (row owners)

  __shared__ alignas(32) float    slds[16][32];
  __shared__ alignas(32) _Float16 plds[16][32];
  __shared__ alignas(32) float    alph[16];

  // K B-fragments: half-1 lanes are pure zero padding; init once, never clobbered
  v16h b0 = {}, b1 = {};

  for (int kb = 0; kb < N_; kb += 32) {
    if (kb + 32 < N_) {
      __builtin_prefetch(Kb + (size_t)(kb + 32) * DPAD + lane * 16, 0, 3);
      __builtin_prefetch(VbT + kb + 32, 0, 3);
    }
    if (half == 0) {                  // exec-masked loads; half-1 keeps zeros
      b0 = *(const v16h*)(Kb + (size_t)(kb + lm) * DPAD);
      b1 = *(const v16h*)(Kb + (size_t)(kb + 16 + lm) * DPAD);
    }
    v8f zc = {};
    v8f s0 = __builtin_amdgcn_wmma_f32_16x16x32_f16(false, aq, false, b0, (short)0, zc, false, false);
    v8f s1 = __builtin_amdgcn_wmma_f32_16x16x32_f16(false, aq, false, b1, (short)0, zc, false, false);

    // dump S tile (C-layout) to LDS
    #pragma unroll
    for (int r = 0; r < 8; ++r) {
      slds[8 * half + r][lm]      = s0[r];
      slds[8 * half + r][16 + lm] = s1[r];
    }
    __syncthreads();

    // row owners: online softmax for one row (vectorized LDS traffic)
    if (lane < 16) {
      v8f r0 = *(const v8f*)&slds[lane][0];
      v8f r1 = *(const v8f*)&slds[lane][8];
      v8f r2 = *(const v8f*)&slds[lane][16];
      v8f r3 = *(const v8f*)&slds[lane][24];
      float m = r0[0];
      #pragma unroll
      for (int e = 0; e < 8; ++e) {
        m = fmaxf(m, r0[e]); m = fmaxf(m, r1[e]);
        m = fmaxf(m, r2[e]); m = fmaxf(m, r3[e]);
      }
      float nm = fmaxf(rmax, m);
      float al = __expf(rmax - nm);
      rmax = nm;
      float t = 0.f;
      v8h p0, p1, p2, p3;
      #pragma unroll
      for (int e = 0; e < 8; ++e) {
        float e0 = __expf(r0[e] - nm); t += e0; p0[e] = (_Float16)e0;
        float e1 = __expf(r1[e] - nm); t += e1; p1[e] = (_Float16)e1;
        float e2 = __expf(r2[e] - nm); t += e2; p2[e] = (_Float16)e2;
        float e3 = __expf(r3[e] - nm); t += e3; p3[e] = (_Float16)e3;
      }
      *(v8h*)&plds[lane][0]  = p0;
      *(v8h*)&plds[lane][8]  = p1;
      *(v8h*)&plds[lane][16] = p2;
      *(v8h*)&plds[lane][24] = p3;
      rsum = rsum * al + t;
      alph[lane] = al;
    }
    __syncthreads();

    // rescale O by this block's alpha (per-row broadcast from LDS)
    v8f al8 = *(const v8f*)&alph[8 * half];
    #pragma unroll
    for (int r = 0; r < 8; ++r) o[r] *= al8[r];

    // P A-fragment: two contiguous 8-half chunks per lane
    v16h pa;
    {
      v8h lo = *(const v8h*)&plds[lm][16 * half];
      v8h hi = *(const v8h*)&plds[lm][16 * half + 8];
      #pragma unroll
      for (int e = 0; e < 8; ++e) { pa[e] = lo[e]; pa[8 + e] = hi[e]; }
    }
    // V B-fragment: transposed layout makes this one contiguous 32B load
    v16h vb = *(const v16h*)(VbT + kb + 16 * half);

    o = __builtin_amdgcn_wmma_f32_16x16x32_f16(false, pa, false, vb, (short)0, o, false, false);
    __syncthreads();
  }

  // broadcast per-row softmax denominators, then write O / rsum
  if (lane < 16) alph[lane] = rsum;
  __syncthreads();
  if (lm < EMB) {
    #pragma unroll
    for (int r = 0; r < 8; ++r) {
      int m = 8 * half + r;
      attn_out[((size_t)b * N_ + (size_t)qt * 16 + m) * 96 + h * EMB + lm] = o[r] / alph[m];
    }
  }
}

// ------------- K6: out@Wo, relu, layernorm -> loc_emb(f16), loc_emb*W(f16)
__global__ __launch_bounds__(32)
void k_locemb(const float* __restrict__ attn_out, const float* __restrict__ Wo,
              const float* __restrict__ W,
              _Float16* __restrict__ loce, _Float16* __restrict__ aw) {
  int row = blockIdx.x;               // b*N + n
  int c = threadIdx.x;                // 0..31 hidden channel
  const float* in = attn_out + (size_t)row * 96;
  float acc = 0.f;
  for (int k = 0; k < 96; ++k) acc += in[k] * Wo[k * HIDDEN + c];
  acc = fmaxf(acc, 0.f);
  float mu = acc;
  #pragma unroll
  for (int off = 1; off < 32; off <<= 1) mu += __shfl_xor(mu, off, 32);
  mu *= (1.0f / 32.0f);
  float d = acc - mu, vv = d * d;
  #pragma unroll
  for (int off = 1; off < 32; off <<= 1) vv += __shfl_xor(vv, off, 32);
  vv *= (1.0f / 32.0f);
  float y = d * rsqrtf(vv + 1e-8f);
  loce[(size_t)row * HIDDEN + c] = (_Float16)y;
  aw[(size_t)row * HIDDEN + c]   = (_Float16)(y * W[c]);
}

// -------- K7: adj tiles (WMMA K=32) + fused per-row top-5 + scatter+norm
//         2 lanes per row scan 8 columns each; 5-candidate merge at the end
__global__ __launch_bounds__(32)
void k_adj_topk(const _Float16* __restrict__ aw, const _Float16* __restrict__ loce,
                float* __restrict__ z) {
  int wg = blockIdx.x;                // b*128 + row-tile
  int rt = wg & 127, b = wg >> 7;
  int lane = threadIdx.x;
  int half = lane >> 4, lm = lane & 15;
  const _Float16* A  = aw   + ((size_t)b * N_ + (size_t)rt * 16) * HIDDEN;
  const _Float16* Bm = loce + (size_t)b * N_ * HIDDEN;

  // A-fragment, K = HIDDEN = 32 exactly: two contiguous 8-half chunks
  v16h af;
  {
    v8h lo = *(const v8h*)(A + lm * HIDDEN + 16 * half);
    v8h hi = *(const v8h*)(A + lm * HIDDEN + 16 * half + 8);
    #pragma unroll
    for (int e = 0; e < 8; ++e) { af[e] = lo[e]; af[8 + e] = hi[e]; }
  }

  float tv[TOPK]; int ti[TOPK];
  #pragma unroll
  for (int t = 0; t < TOPK; ++t) { tv[t] = -1.f; ti[t] = t; }

  __shared__ alignas(32) float tile[16][17];
  __shared__ float mv[16][TOPK];
  __shared__ int   mi[16][TOPK];

  int cbase = half * 8;               // this lane scans columns [cbase, cbase+8)

  for (int ct = 0; ct < 128; ++ct) {
    if (ct + 1 < 128) __builtin_prefetch(Bm + (size_t)(ct + 1) * 16 * HIDDEN + lane * 16, 0, 3);
    // B-fragment: one contiguous 16-half (32B) chunk per lane
    v16h bf = *(const v16h*)(Bm + (size_t)(ct * 16 + lm) * HIDDEN + 16 * half);
    v8f zc = {};
    v8f c = __builtin_amdgcn_wmma_f32_16x16x32_f16(false, af, false, bf, (short)0, zc, false, false);
    #pragma unroll
    for (int r = 0; r < 8; ++r) tile[8 * half + r][lm] = fmaxf(c[r], 0.f);
    __syncthreads();
    #pragma unroll
    for (int cc = 0; cc < 8; ++cc) {  // all 32 lanes active: row lm, col half*8+cc
      float v = tile[lm][cbase + cc];
      int idx = ct * 16 + cbase + cc;
      #pragma unroll
      for (int t = 0; t < TOPK; ++t) {     // branchless sorted insert
        bool gt = v > tv[t];
        float ov = tv[t]; int oi = ti[t];
        tv[t] = gt ? v : ov;  ti[t] = gt ? idx : oi;
        v = gt ? ov : v;      idx = gt ? oi : idx;
      }
    }
    __syncthreads();
  }
  // merge the two per-row candidate lists
  if (lane >= 16) {
    #pragma unroll
    for (int t = 0; t < TOPK; ++t) { mv[lm][t] = tv[t]; mi[lm][t] = ti[t]; }
  }
  __syncthreads();
  if (lane < 16) {
    #pragma unroll
    for (int t0 = 0; t0 < TOPK; ++t0) {
      float v = mv[lm][t0]; int idx = mi[lm][t0];
      #pragma unroll
      for (int t = 0; t < TOPK; ++t) {
        bool gt = v > tv[t];
        float ov = tv[t]; int oi = ti[t];
        tv[t] = gt ? v : ov;  ti[t] = gt ? idx : oi;
        v = gt ? ov : v;      idx = gt ? oi : idx;
      }
    }
    float s = 0.f;
    #pragma unroll
    for (int t = 0; t < TOPK; ++t) s += fmaxf(tv[t], 0.f);
    float inv = 1.0f / (s + 1e-5f);
    size_t rb = ((size_t)b * N_ + (size_t)rt * 16 + lm) * N_;
    #pragma unroll
    for (int t = 0; t < TOPK; ++t) z[rb + ti[t]] = fmaxf(tv[t], 0.f) * inv;
  }
}

// ---------------------------------------------------------------- launcher
extern "C" void kernel_launch(void* const* d_in, const int* in_sizes, int n_in,
                              void* d_out, int out_size, void* d_ws, size_t ws_size,
                              hipStream_t stream) {
  const float* x   = (const float*)d_in[0];
  const float* loc = (const float*)d_in[2];
  const float* td  = (const float*)d_in[3];
  const float* wq  = (const float*)d_in[5];
  const float* wk  = (const float*)d_in[6];
  const float* wv  = (const float*)d_in[7];
  const float* Wq  = (const float*)d_in[8];
  const float* Wk  = (const float*)d_in[9];
  const float* Wv  = (const float*)d_in[10];
  const float* Wo  = (const float*)d_in[11];
  const float* W   = (const float*)d_in[12];
  float* z = (float*)d_out;

  char* ws = (char*)d_ws;
  size_t off = 0;
  auto carve = [&](size_t bytes) -> void* {
    void* p = ws + off;
    off += (bytes + 255) & ~(size_t)255;
    return p;
  };
  const size_t ROWS = (size_t)B_ * N_;
  _Float16* xh   = (_Float16*)carve(ROWS * T_ * 2);
  _Float16* WcT  = (_Float16*)carve((size_t)FPAD * T_ * 2);
  _Float16* WsT  = (_Float16*)carve((size_t)FPAD * T_ * 2);
  float*    mag  = (float*)   carve(ROWS * FPAD * 4);
  float*    xn   = (float*)   carve(ROWS * DX * 4);
  float*    tl   = (float*)   carve(ROWS * DY * 4);
  float*    Cq   = (float*)   carve((size_t)DY * 96 * 4);
  float*    Ck   = (float*)   carve((size_t)DX * 96 * 4);
  float*    Cv   = (float*)   carve((size_t)DX * 96 * 4);
  _Float16* qh   = (_Float16*)carve((size_t)B_ * NHEAD * N_ * DPAD * 2);
  _Float16* kh   = (_Float16*)carve((size_t)B_ * NHEAD * N_ * DPAD * 2);
  _Float16* vhT  = (_Float16*)carve((size_t)B_ * NHEAD * N_ * DPAD * 2);
  float*    aout = (float*)   carve(ROWS * 96 * 4);
  _Float16* loce = (_Float16*)carve(ROWS * HIDDEN * 2);
  _Float16* awb  = (_Float16*)carve(ROWS * HIDDEN * 2);
  (void)ws_size; (void)in_sizes; (void)n_in;

  // z is sparse (top-5 per row): zero it, then scatter
  hipMemsetAsync(d_out, 0, (size_t)out_size * sizeof(float), stream);

  k_tof16   <<<(int)((ROWS * T_ + 255) / 256), 256, 0, stream>>>(x, xh, (int)(ROWS * T_));
  k_twiddle <<<(FPAD * T_ + 255) / 256, 256, 0, stream>>>(WcT, WsT);
  k_dft     <<<(int)(ROWS / 16) * (FPAD / 16), 32, 0, stream>>>(xh, WcT, WsT, mag);
  k_specnorm<<<(int)ROWS, 256, 0, stream>>>(mag, xn);
  k_tdloc   <<<(int)ROWS, 256, 0, stream>>>(loc, td, tl);
  k_combine <<<(DY * 96 + 255) / 256, 256, 0, stream>>>(wq, Wq, Cq, DY);
  k_combine <<<(DX * 96 + 255) / 256, 256, 0, stream>>>(wk, Wk, Ck, DX);
  k_combine <<<(DX * 96 + 255) / 256, 256, 0, stream>>>(wv, Wv, Cv, DX);
  k_qkv     <<<(int)ROWS, 128, 0, stream>>>(tl, xn, Cq, Ck, Cv, qh, kh, vhT);
  k_attn    <<<B_ * NHEAD * (N_ / 16), 32, 0, stream>>>(qh, kh, vhT, aout);
  k_locemb  <<<(int)ROWS, 32, 0, stream>>>(aout, Wo, W, loce, awb);
  k_adj_topk<<<B_ * (N_ / 16), 32, 0, stream>>>(awb, loce, z);
}
